// OA_0_14628658610726
// MI455X (gfx1250) — matrix operand order repaired and checked by reading, not compiled, not run on hardware
//
#include <hip/hip_runtime.h>
#include <hip/hip_bf16.h>

#define B_DIM 8
#define C_DIM 256
#define N_DIM 3072
#define ELD (N_DIM + 4)   // fp32 energy row stride in LDS (bank-spread)
#define ALD (N_DIM + 8)   // bf16 attention row stride in LDS (16B aligned)

typedef unsigned short u16;
typedef __bf16 bf16_t;
typedef bf16_t v16bf __attribute__((ext_vector_type(16)));
typedef float  v8f   __attribute__((ext_vector_type(8)));

static __device__ __forceinline__ u16 f2bf(float f) {
  unsigned u = __float_as_uint(f);
  u += 0x7fffu + ((u >> 16) & 1u);       // round-to-nearest-even
  return (u16)(u >> 16);
}
static __device__ __forceinline__ float bf2f(u16 h) {
  return __uint_as_float(((unsigned)h) << 16);
}

static __device__ __forceinline__ v8f wmma_bf16(v16bf a, v16bf b, v8f c) {
  // D(f32 16x16) = A(16x32 bf16) * B(32x16 bf16) + C
  return __builtin_amdgcn_wmma_f32_16x16x32_bf16(false, a, false, b, (short)0, c,
                                                 false, false);
}

// A fragment (16x32 bf16) from row-major [M x K], lda = row stride in elems.
// ISA layout: lanes 0-15 rows 0-15 K[0..7]/K[16..23], lanes 16-31 K[8..15]/K[24..31]
static __device__ __forceinline__ v16bf load_frag_A(const u16* base, int lda,
                                                    int row, int k, int lane) {
  int m = lane & 15, hi = lane >> 4;
  const u16* p = base + (size_t)(row + m) * lda + (k + hi * 8);
  union { uint4 u[2]; v16bf v; } U;
  U.u[0] = *(const uint4*)(p);
  U.u[1] = *(const uint4*)(p + 16);
  return U.v;
}
// B fragment (32x16 bf16) from Bt row-major [Ncol x K] (i.e. B transposed).
// ISA layout: lanes 0-15 col=lane K[0..7]/K[16..23]; lanes 16-31 col=lane-16 K[8..15]/K[24..31]
static __device__ __forceinline__ v16bf load_frag_B(const u16* base, int ldb,
                                                    int col, int k, int lane) {
  int n = lane & 15, hi = lane >> 4;
  const u16* p = base + (size_t)(col + n) * ldb + (k + hi * 8);
  union { uint4 u[2]; v16bf v; } U;
  U.u[0] = *(const uint4*)(p);
  U.u[1] = *(const uint4*)(p + 16);
  return U.v;
}

// ---------------- prep kernels ----------------
__global__ void cvt_bf16_kernel(const float* __restrict__ s, u16* __restrict__ d, int n) {
  int i = blockIdx.x * blockDim.x + threadIdx.x;
  if (i < n) d[i] = f2bf(s[i]);
}

// [B,C,N] f32 -> [B,N,C] bf16 via LDS tile transpose
__global__ __launch_bounds__(256) void transpose_cvt_kernel(const float* __restrict__ src,
                                                            u16* __restrict__ dst) {
  __shared__ float tile[32][33];
  int b = blockIdx.z, c0 = blockIdx.y * 32, n0 = blockIdx.x * 32;
  int tx = threadIdx.x, ty = threadIdx.y;
  const float* s = src + ((size_t)b * C_DIM + c0) * N_DIM + n0;
  for (int i = ty; i < 32; i += 8) tile[i][tx] = s[(size_t)i * N_DIM + tx];
  __syncthreads();
  u16* d = dst + ((size_t)b * N_DIM + n0) * C_DIM + c0;
  for (int i = ty; i < 32; i += 8) d[(size_t)i * C_DIM + tx] = f2bf(tile[tx][i]);
}

// ---------------- generic WMMA GEMM: D[M,Nc] = A[M,K] x Bt[Nc,K]^T ----------------
// one wave computes a 16x64 stripe (A fragment reused x4, 4 independent WMMA chains)
template <bool OUT_BF16, bool BIAS>
__global__ __launch_bounds__(128) void gemm_nt_kernel(
    const u16* __restrict__ A, long long strideA, int lda,
    const u16* __restrict__ Bt, long long strideB, int ldb,
    void* __restrict__ Out, long long strideO, int ldo,
    const float* __restrict__ bias, int K) {
  int lane = threadIdx.x & 31, wave = threadIdx.x >> 5;
  int b = blockIdx.z;
  int rowTile = blockIdx.y * 16;
  int colGroup = (blockIdx.x * 4 + wave) * 64;
  const u16* Ab = A + (size_t)b * strideA;
  const u16* Bb = Bt + (size_t)b * strideB;
  const v8f z = {0.f, 0.f, 0.f, 0.f, 0.f, 0.f, 0.f, 0.f};
  v8f acc[4] = {z, z, z, z};
  for (int k = 0; k < K; k += 32) {
    v16bf a = load_frag_A(Ab, lda, rowTile, k, lane);
#pragma unroll
    for (int t = 0; t < 4; ++t) {
      v16bf bf = load_frag_B(Bb, ldb, colGroup + t * 16, k, lane);
      acc[t] = wmma_bf16(a, bf, acc[t]);
    }
  }
  int rlo = (lane >> 4) * 8, cn = lane & 15;
#pragma unroll
  for (int t = 0; t < 4; ++t) {
    int col = colGroup + t * 16 + cn;
#pragma unroll
    for (int j = 0; j < 8; ++j) {
      int row = rowTile + rlo + j;
      float v = acc[t][j];
      if (BIAS) v += bias[row];
      if (OUT_BF16)
        ((u16*)Out)[(size_t)b * strideO + (size_t)row * ldo + col] = f2bf(v);
      else
        ((float*)Out)[(size_t)b * strideO + (size_t)row * ldo + col] = v;
    }
  }
}

// ---------------- energy row-block: 16 rows x 3072 cols into LDS ----------------
// two column tiles per iteration -> two independent WMMA accumulation chains
static __device__ __forceinline__ void energy_rowblock(const u16* __restrict__ Qb,
                                                       const u16* __restrict__ Kb,
                                                       int rowTile, float* sE,
                                                       int lane, int wave) {
  const v8f z = {0.f, 0.f, 0.f, 0.f, 0.f, 0.f, 0.f, 0.f};
  v16bf afr[8];  // Q rows for this block: 16x256 bf16, kept in VGPRs, reused 24x
#pragma unroll
  for (int ks = 0; ks < 8; ++ks) afr[ks] = load_frag_A(Qb, C_DIM, rowTile, ks * 32, lane);
  int r0 = (lane >> 4) * 8, mlo = lane & 15;
  for (int ct = wave; ct < N_DIM / 16; ct += 16) {
    int ct2 = ct + 8;
    v8f acc0 = z, acc1 = z;
#pragma unroll
    for (int ks = 0; ks < 8; ++ks) {
      v16bf b0 = load_frag_B(Kb, C_DIM, ct * 16, ks * 32, lane);
      v16bf b1 = load_frag_B(Kb, C_DIM, ct2 * 16, ks * 32, lane);
      acc0 = wmma_bf16(afr[ks], b0, acc0);
      acc1 = wmma_bf16(afr[ks], b1, acc1);
    }
    int m0 = ct * 16 + mlo, m1 = ct2 * 16 + mlo;
#pragma unroll
    for (int j = 0; j < 8; ++j) {
      sE[(r0 + j) * ELD + m0] = acc0[j];
      sE[(r0 + j) * ELD + m1] = acc1[j];
    }
  }
}

// ---------------- K2: energy -> row softmax stats + column-sum accumulation ----------------
__global__ __launch_bounds__(256) void energy_softmax_kernel(
    const u16* __restrict__ Qm, const u16* __restrict__ Km,
    float* __restrict__ colsum, float* __restrict__ rowmax, float* __restrict__ rowinv) {
  extern __shared__ float sE[];  // 16 x ELD fp32
  __shared__ float red[256];
  __shared__ float rmS[16], riS[16];
  int b = blockIdx.y, rowTile = blockIdx.x * 16;
  int lane = threadIdx.x & 31, wave = threadIdx.x >> 5;
  const u16* Qb = Qm + (size_t)b * N_DIM * C_DIM;
  const u16* Kb = Km + (size_t)b * N_DIM * C_DIM;
  energy_rowblock(Qb, Kb, rowTile, sE, lane, wave);
  __syncthreads();

  int r = threadIdx.x >> 4, sub = threadIdx.x & 15;
  const int CH = N_DIM / 16;  // 192 cols per thread-chunk
  float mx = -3.4e38f;
  for (int i = 0; i < CH; ++i) mx = fmaxf(mx, sE[r * ELD + sub * CH + i]);
  red[threadIdx.x] = mx;
  __syncthreads();
  float rmax = -3.4e38f;
  for (int i = 0; i < 16; ++i) rmax = fmaxf(rmax, red[r * 16 + i]);
  __syncthreads();
  float ps = 0.f;
  for (int i = 0; i < CH; ++i) ps += __expf(sE[r * ELD + sub * CH + i] - rmax);
  red[threadIdx.x] = ps;
  __syncthreads();
  float rsum = 0.f;
  for (int i = 0; i < 16; ++i) rsum += red[r * 16 + i];
  float rinv = 1.f / rsum;
  if (sub == 0) {
    rmS[r] = rmax;
    riS[r] = rinv;
    rowmax[(size_t)b * N_DIM + rowTile + r] = rmax;
    rowinv[(size_t)b * N_DIM + rowTile + r] = rinv;
  }
  __syncthreads();
  for (int m = threadIdx.x; m < N_DIM; m += 256) {
    float s = 0.f;
    for (int rr = 0; rr < 16; ++rr) s += __expf(sE[rr * ELD + m] - rmS[rr]) * riS[rr];
    atomicAdd(&colsum[(size_t)b * N_DIM + m], s);
  }
}

// ---------------- K3: recompute energy, normalized attention (LDS bf16), A*V, d = q - x_r ----------------
__global__ __launch_bounds__(256) void attn_apply_kernel(
    const u16* __restrict__ Qm, const u16* __restrict__ Km,
    const u16* __restrict__ Vt, const u16* __restrict__ qT,
    const float* __restrict__ colsum, const float* __restrict__ rowmax,
    const float* __restrict__ rowinv, u16* __restrict__ dbf) {
  extern __shared__ char smem[];
  float* sE = (float*)smem;                                       // 16 x ELD f32
  u16* sAtt = (u16*)(smem + (size_t)16 * ELD * sizeof(float));    // 16 x ALD bf16
  __shared__ float rmS[16], riS[16];
  int b = blockIdx.y, rowTile = blockIdx.x * 16;
  int lane = threadIdx.x & 31, wave = threadIdx.x >> 5;
  const u16* Qb = Qm + (size_t)b * N_DIM * C_DIM;
  const u16* Kb = Km + (size_t)b * N_DIM * C_DIM;
  const u16* Vb = Vt + (size_t)b * C_DIM * N_DIM;
  energy_rowblock(Qb, Kb, rowTile, sE, lane, wave);
  if (threadIdx.x < 16) {
    rmS[threadIdx.x] = rowmax[(size_t)b * N_DIM + rowTile + threadIdx.x];
    riS[threadIdx.x] = rowinv[(size_t)b * N_DIM + rowTile + threadIdx.x];
  }
  __syncthreads();

  // normalized attention -> LDS bf16 (column reciprocal hoisted: m outer, r inner)
  const float* csb = colsum + (size_t)b * N_DIM;
  for (int m = threadIdx.x; m < N_DIM; m += 256) {
    float rs = 1.0f / (1e-9f + csb[m]);   // column L1 renorm
#pragma unroll
    for (int r = 0; r < 16; ++r) {
      float a = __expf(sE[r * ELD + m] - rmS[r]) * (riS[r] * rs);  // row softmax * renorm
      sAtt[r * ALD + m] = f2bf(a);
    }
  }
  __syncthreads();

  // x_r tile GEMM: D[16 x 32] per wave, K = 3072, A from LDS, B = Vt rows (global)
  const v8f z = {0.f, 0.f, 0.f, 0.f, 0.f, 0.f, 0.f, 0.f};
  int c0 = wave * 32;
  v8f acc0 = z, acc1 = z;
  for (int ks = 0; ks < N_DIM / 32; ++ks) {
    v16bf a = load_frag_A(sAtt, ALD, 0, ks * 32, lane);
    v16bf b0 = load_frag_B(Vb, N_DIM, c0, ks * 32, lane);
    v16bf b1 = load_frag_B(Vb, N_DIM, c0 + 16, ks * 32, lane);
    acc0 = wmma_bf16(a, b0, acc0);
    acc1 = wmma_bf16(a, b1, acc1);
  }
  int rlo = (lane >> 4) * 8, cn = lane & 15;
  const u16* qTb = qT + (size_t)b * N_DIM * C_DIM;
  u16* db = dbf + (size_t)b * N_DIM * C_DIM;
#pragma unroll
  for (int j = 0; j < 8; ++j) {
    int n = rowTile + rlo + j;
    int ca = c0 + cn, cb = c0 + 16 + cn;
    db[(size_t)n * C_DIM + ca] = f2bf(bf2f(qTb[(size_t)n * C_DIM + ca]) - acc0[j]);
    db[(size_t)n * C_DIM + cb] = f2bf(bf2f(qTb[(size_t)n * C_DIM + cb]) - acc1[j]);
  }
}

// ---------------- BN stats: per channel over (B, N) ----------------
__global__ __launch_bounds__(256) void bn_stats_kernel(const float* __restrict__ tb,
                                                       float* __restrict__ meanv,
                                                       float* __restrict__ rstdv) {
  int o = blockIdx.x;
  float s = 0.f, s2 = 0.f;
  for (int b = 0; b < B_DIM; ++b) {
    const float* p = tb + ((size_t)b * C_DIM + o) * N_DIM;
    for (int n = threadIdx.x; n < N_DIM; n += 256) {
      float v = p[n];
      s += v;
      s2 += v * v;
    }
  }
  __shared__ float rs[256], rs2[256];
  rs[threadIdx.x] = s;
  rs2[threadIdx.x] = s2;
  __syncthreads();
  for (int off = 128; off; off >>= 1) {
    if (threadIdx.x < off) {
      rs[threadIdx.x] += rs[threadIdx.x + off];
      rs2[threadIdx.x] += rs2[threadIdx.x + off];
    }
    __syncthreads();
  }
  if (threadIdx.x == 0) {
    const float cnt = (float)B_DIM * (float)N_DIM;
    float m = rs[0] / cnt;
    float var = rs2[0] / cnt - m * m;
    meanv[o] = m;
    rstdv[o] = rsqrtf(var + 1e-5f);
  }
}

// ---------------- fused BN + ReLU + residual ----------------
__global__ void final_kernel(const float* __restrict__ q, const float* __restrict__ tb,
                             const float* __restrict__ meanv, const float* __restrict__ rstdv,
                             const float* __restrict__ gamma, const float* __restrict__ beta,
                             float* __restrict__ out) {
  int i = blockIdx.x * blockDim.x + threadIdx.x;  // < B*C*N = 6291456
  int c = (i / N_DIM) & (C_DIM - 1);
  float h = (tb[i] - meanv[c]) * rstdv[c] * gamma[c] + beta[c];
  out[i] = q[i] + fmaxf(h, 0.f);
}

extern "C" void kernel_launch(void* const* d_in, const int* in_sizes, int n_in,
                              void* d_out, int out_size, void* d_ws, size_t ws_size,
                              hipStream_t stream) {
  (void)in_sizes; (void)n_in; (void)out_size; (void)ws_size;
  const float* x  = (const float*)d_in[0];
  const float* q  = (const float*)d_in[1];
  const float* Wq = (const float*)d_in[2];
  const float* Wk = (const float*)d_in[3];
  const float* Wv = (const float*)d_in[4];
  const float* bv = (const float*)d_in[5];
  const float* Wt = (const float*)d_in[6];
  const float* bt = (const float*)d_in[7];
  const float* gamma = (const float*)d_in[8];
  const float* beta  = (const float*)d_in[9];
  float* out = (float*)d_out;

  char* ws = (char*)d_ws;
  size_t off = 0;
  auto alloc = [&](size_t bytes) -> void* {
    void* p = ws + off;
    off += (bytes + 255) & ~(size_t)255;
    return p;
  };
  const size_t CC = (size_t)C_DIM * C_DIM;
  const size_t BNC = (size_t)B_DIM * N_DIM * C_DIM;
  u16* wq_bf = (u16*)alloc(CC * 2);
  u16* wk_bf = (u16*)alloc(CC * 2);
  u16* wv_bf = (u16*)alloc(CC * 2);
  u16* wt_bf = (u16*)alloc(CC * 2);
  u16* qT   = (u16*)alloc(BNC * 2);   // q^T  [B,N,C] bf16
  u16* xT   = (u16*)alloc(BNC * 2);   // x^T  [B,N,C] bf16
  u16* Qm   = (u16*)alloc(BNC * 2);   // x_q  [B,N,C] bf16
  u16* Km   = (u16*)alloc(BNC * 2);   // x_k^T [B,N,C] bf16
  u16* Vt   = (u16*)alloc(BNC * 2);   // x_v^T as [B,C,N] bf16
  u16* dbf  = (u16*)alloc(BNC * 2);   // (q - x_r)^T [B,N,C] bf16
  float* tbuf   = (float*)alloc(BNC * 4);             // t [B,C,N] f32
  float* colsum = (float*)alloc((size_t)B_DIM * N_DIM * 4);
  float* rowmax = (float*)alloc((size_t)B_DIM * N_DIM * 4);
  float* rowinv = (float*)alloc((size_t)B_DIM * N_DIM * 4);
  float* meanv  = (float*)alloc(C_DIM * 4);
  float* rstdv  = (float*)alloc(C_DIM * 4);

  // weights -> bf16
  cvt_bf16_kernel<<<(int)(CC / 256), 256, 0, stream>>>(Wq, wq_bf, (int)CC);
  cvt_bf16_kernel<<<(int)(CC / 256), 256, 0, stream>>>(Wk, wk_bf, (int)CC);
  cvt_bf16_kernel<<<(int)(CC / 256), 256, 0, stream>>>(Wv, wv_bf, (int)CC);
  cvt_bf16_kernel<<<(int)(CC / 256), 256, 0, stream>>>(Wt, wt_bf, (int)CC);
  // activations -> transposed bf16
  dim3 tb(32, 8), tg(N_DIM / 32, C_DIM / 32, B_DIM);
  transpose_cvt_kernel<<<tg, tb, 0, stream>>>(q, qT);
  transpose_cvt_kernel<<<tg, tb, 0, stream>>>(x, xT);
  hipMemsetAsync(colsum, 0, (size_t)B_DIM * N_DIM * 4, stream);

  const long long sBNC = (long long)N_DIM * C_DIM;
  const long long sBCN = (long long)C_DIM * N_DIM;
  // x_q[n,o]: A=qT, Bt=Wq (row o, k c)
  gemm_nt_kernel<true, false><<<dim3(1, N_DIM / 16, B_DIM), 128, 0, stream>>>(
      qT, sBNC, C_DIM, wq_bf, 0, C_DIM, Qm, sBNC, C_DIM, nullptr, C_DIM);
  // x_k^T[m,o]: A=xT, Bt=Wk
  gemm_nt_kernel<true, false><<<dim3(1, N_DIM / 16, B_DIM), 128, 0, stream>>>(
      xT, sBNC, C_DIM, wk_bf, 0, C_DIM, Km, sBNC, C_DIM, nullptr, C_DIM);
  // x_v^T stored [C,N]: A=Wv, Bt=xT, bias bv per row o
  gemm_nt_kernel<true, true><<<dim3(N_DIM / 64 / 4, C_DIM / 16, B_DIM), 128, 0, stream>>>(
      wv_bf, 0, C_DIM, xT, sBNC, C_DIM, Vt, sBCN, N_DIM, bv, C_DIM);

  // K2: softmax stats + colsum (197KB LDS)
  const int EBYTES = 16 * ELD * 4;
  hipFuncSetAttribute((const void*)energy_softmax_kernel,
                      hipFuncAttributeMaxDynamicSharedMemorySize, EBYTES);
  energy_softmax_kernel<<<dim3(N_DIM / 16, B_DIM), 256, EBYTES, stream>>>(
      Qm, Km, colsum, rowmax, rowinv);

  // K3: recompute + normalize + A*V + residual diff (295KB LDS)
  const int K3BYTES = EBYTES + 16 * ALD * 2;
  hipFuncSetAttribute((const void*)attn_apply_kernel,
                      hipFuncAttributeMaxDynamicSharedMemorySize, K3BYTES);
  attn_apply_kernel<<<dim3(N_DIM / 16, B_DIM), 256, K3BYTES, stream>>>(
      Qm, Km, Vt, qT, colsum, rowmax, rowinv, dbf);

  // t[o,n] = Wt * d^T + bt  (f32 out)
  gemm_nt_kernel<false, true><<<dim3(N_DIM / 64 / 4, C_DIM / 16, B_DIM), 128, 0, stream>>>(
      wt_bf, 0, C_DIM, dbf, sBNC, C_DIM, tbuf, sBCN, N_DIM, bt, C_DIM);

  bn_stats_kernel<<<C_DIM, 256, 0, stream>>>(tbuf, meanv, rstdv);
  final_kernel<<<(B_DIM * C_DIM * N_DIM) / 256, 256, 0, stream>>>(
      q, tbuf, meanv, rstdv, gamma, beta, out);
}
